// GNNModel_23802708754824
// MI455X (gfx1250) — compile-verified, matrix operand order checked by reading, not run on hardware
//
#include <hip/hip_runtime.h>
#include <hip/hip_bf16.h>

// ---------------- problem constants ----------------
#define NN 50000      // nodes
#define NE 800000     // edges
#define DIN 128
#define DH  256
#define NG  256       // graphs
#define NC  2         // classes

typedef __attribute__((ext_vector_type(8)))  float  v8f;
typedef __attribute__((ext_vector_type(8)))  __bf16 v8bf;
typedef __attribute__((ext_vector_type(16))) __bf16 v16bf;

// ---------------- bf16 helpers (bit-level, storage = unsigned short) --------
__device__ __forceinline__ float bf2f(unsigned short b) {
    unsigned int u = ((unsigned int)b) << 16;
    float f; __builtin_memcpy(&f, &u, 4); return f;
}
__device__ __forceinline__ unsigned short f2bf(float f) {
    unsigned int u; __builtin_memcpy(&u, &f, 4);
    unsigned int r = u + 0x7FFFu + ((u >> 16) & 1u);   // round-to-nearest-even
    return (unsigned short)(r >> 16);
}

// ---------------- f32 -> bf16 bulk convert ----------------
__global__ __launch_bounds__(256) void k_f2bf(const float* __restrict__ in,
                                              unsigned short* __restrict__ out, int n) {
    int i = blockIdx.x * 256 + threadIdx.x;
    if (i < n) out[i] = f2bf(in[i]);
}

// ---------------- weight pack: [Wroot;Wrel] (K=2*Ksrc, N=256) into per-lane
// B-fragment layout: Wpack[((nt*ktiles+kt)*32+lane)*16 + j] ----------------
__global__ __launch_bounds__(256) void k_pack(const float* __restrict__ Wroot,
                                              const float* __restrict__ Wrel,
                                              int Ksrc, unsigned short* __restrict__ Wpack) {
    int ktiles = (Ksrc * 2) >> 5;
    int total  = 16 * ktiles * 512;
    int idx = blockIdx.x * 256 + threadIdx.x;
    if (idx >= total) return;
    int j    = idx & 15;
    int lane = (idx >> 4) & 31;
    int rest = idx >> 9;
    int kt   = rest % ktiles;
    int nt   = rest / ktiles;
    int koff = (lane >> 4) * 8;
    int k    = kt * 32 + ((j < 8) ? (koff + j) : (16 + koff + (j - 8)));
    int n    = nt * 16 + (lane & 15);
    float v  = (k < Ksrc) ? Wroot[(size_t)n * Ksrc + k]
                          : Wrel [(size_t)n * Ksrc + (k - Ksrc)];
    Wpack[idx] = f2bf(v);
}

// ---------------- edge scatter-add, f32 source (layer 1, D=128) ----------------
__global__ __launch_bounds__(256) void k_scatter_f32(const float* __restrict__ x,
                                                     const long long* __restrict__ ei,
                                                     float* __restrict__ agg, int nE) {
    int t = blockIdx.x * 256 + threadIdx.x;
    int e = t >> 5; if (e >= nE) return;
    int lane = t & 31;
    long long s = ei[e];
    long long d = ei[nE + e];
    const float4 v = *(const float4*)(x + (size_t)s * DIN + lane * 4);
    float* dst = agg + (size_t)d * DIN + lane * 4;
    atomicAdd(dst + 0, v.x); atomicAdd(dst + 1, v.y);
    atomicAdd(dst + 2, v.z); atomicAdd(dst + 3, v.w);
}

// ---------------- edge scatter-add, bf16 source (layer 2, D=256) ----------------
__global__ __launch_bounds__(256) void k_scatter_bf(const unsigned short* __restrict__ h,
                                                    const long long* __restrict__ ei,
                                                    float* __restrict__ agg, int nE) {
    int t = blockIdx.x * 256 + threadIdx.x;
    int e = t >> 5; if (e >= nE) return;
    int lane = t & 31;
    long long s = ei[e];
    long long d = ei[nE + e];
    const uint4 v = *(const uint4*)(h + (size_t)s * DH + lane * 8);
    float* dst = agg + (size_t)d * DH + lane * 8;
    unsigned int w[4] = {v.x, v.y, v.z, v.w};
#pragma unroll
    for (int q = 0; q < 4; ++q) {
        atomicAdd(dst + q * 2 + 0, bf2f((unsigned short)(w[q] & 0xFFFFu)));
        atomicAdd(dst + q * 2 + 1, bf2f((unsigned short)(w[q] >> 16)));
    }
}

// ---------------- WMMA fragment helpers ----------------
__device__ __forceinline__ v16bf load_afrag(const unsigned short* p /* row base + kbase + koff */) {
    v8bf lo = *(const v8bf*)(p);
    v8bf hi = *(const v8bf*)(p + 16);
    union { v16bf v; v8bf h[2]; } u;
    u.h[0] = lo; u.h[1] = hi;
    return u.v;
}

// ---------------- fused GraphConv GEMM: out = relu([A0|A1] @ Wcat^T + bias) ------
// A0: [M, D0] bf16, A1: [M, D1] bf16 (row-major). N fixed = 256.
// Block = 8 waves, one 16-row M-tile; wave w handles N-tiles 2w, 2w+1.
// K-loop is split into an A0 phase and an A1 phase so the hot loop carries no
// per-iteration source selection (pure loads + WMMA).
template <bool OUT_F32>
__global__ __launch_bounds__(256) void k_gemm(const unsigned short* __restrict__ A0,
                                              const unsigned short* __restrict__ A1,
                                              int D0, int D1,
                                              const unsigned short* __restrict__ Wpack,
                                              const float* __restrict__ bias,
                                              void* __restrict__ outp) {
    const int mbase = blockIdx.x * 16;
    const int wave  = threadIdx.x >> 5;
    const int lane  = threadIdx.x & 31;
    const int row   = mbase + (lane & 15);
    const int koff  = (lane >> 4) * 8;
    const int kt0   = D0 >> 5;
    const int kt1   = D1 >> 5;
    const int ktiles = kt0 + kt1;
    const int nt0 = wave * 2, nt1 = wave * 2 + 1;

    const unsigned short* a0row = A0 + (size_t)row * D0 + koff;
    const unsigned short* a1row = A1 + (size_t)row * D1 + koff;
    const unsigned short* bp0   = Wpack + ((size_t)(nt0 * ktiles) * 32 + lane) * 16;
    const unsigned short* bp1   = Wpack + ((size_t)(nt1 * ktiles) * 32 + lane) * 16;

    v8f acc0 = {}; v8f acc1 = {};

    // phase 1: A0 (root features)
    for (int kt = 0; kt < kt0; ++kt) {
        v16bf a  = load_afrag(a0row + kt * 32);
        v16bf b0 = *(const v16bf*)(bp0 + (size_t)kt * 512);
        v16bf b1 = *(const v16bf*)(bp1 + (size_t)kt * 512);
        acc0 = __builtin_amdgcn_wmma_f32_16x16x32_bf16(false, a, false, b0,
                                                       (short)0, acc0, false, false);
        acc1 = __builtin_amdgcn_wmma_f32_16x16x32_bf16(false, a, false, b1,
                                                       (short)0, acc1, false, false);
    }
    // phase 2: A1 (aggregated neighbor features)
    const unsigned short* bq0 = bp0 + (size_t)kt0 * 512;
    const unsigned short* bq1 = bp1 + (size_t)kt0 * 512;
    for (int kt = 0; kt < kt1; ++kt) {
        v16bf a  = load_afrag(a1row + kt * 32);
        v16bf b0 = *(const v16bf*)(bq0 + (size_t)kt * 512);
        v16bf b1 = *(const v16bf*)(bq1 + (size_t)kt * 512);
        acc0 = __builtin_amdgcn_wmma_f32_16x16x32_bf16(false, a, false, b0,
                                                       (short)0, acc0, false, false);
        acc1 = __builtin_amdgcn_wmma_f32_16x16x32_bf16(false, a, false, b1,
                                                       (short)0, acc1, false, false);
    }

    // epilogue: D layout — VGPR r: lanes 0-15 -> M=r, lanes 16-31 -> M=r+8; N=lane&15
    const int colL = lane & 15;
    const int rofs = (lane >> 4) * 8;
    const int col0 = nt0 * 16 + colL;
    const int col1 = nt1 * 16 + colL;
    const float bc0 = bias[col0];
    const float bc1 = bias[col1];
#pragma unroll
    for (int r = 0; r < 8; ++r) {
        size_t rr = (size_t)(mbase + rofs + r) * DH;
        float v0 = acc0[r] + bc0; v0 = v0 > 0.f ? v0 : 0.f;
        float v1 = acc1[r] + bc1; v1 = v1 > 0.f ? v1 : 0.f;
        if (OUT_F32) {
            ((float*)outp)[rr + col0] = v0;
            ((float*)outp)[rr + col1] = v1;
        } else {
            ((unsigned short*)outp)[rr + col0] = f2bf(v0);
            ((unsigned short*)outp)[rr + col1] = f2bf(v1);
        }
    }
}

// ---------------- per-graph counts ----------------
__global__ __launch_bounds__(256) void k_counts(const long long* __restrict__ batch,
                                                float* __restrict__ counts, int n) {
    int i = blockIdx.x * 256 + threadIdx.x;
    if (i < n) atomicAdd(&counts[(int)batch[i]], 1.0f);
}

// ---------------- segment sum pooling ----------------
__global__ __launch_bounds__(256) void k_pool(const float* __restrict__ h2,
                                              const long long* __restrict__ batch,
                                              float* __restrict__ pooled, int n) {
    int t = blockIdx.x * 256 + threadIdx.x;
    int node = t >> 5; if (node >= n) return;
    int lane = t & 31;
    int g = (int)batch[node];
    const float* src = h2 + (size_t)node * DH + lane * 8;
    float* dst = pooled + (size_t)g * DH + lane * 8;
#pragma unroll
    for (int j = 0; j < 8; ++j) atomicAdd(dst + j, src[j]);
}

// ---------------- head: mean, fc, sigmoid ----------------
__global__ __launch_bounds__(256) void k_head(const float* __restrict__ pooled,
                                              const float* __restrict__ counts,
                                              const float* __restrict__ Wfc,
                                              const float* __restrict__ bfc,
                                              float* __restrict__ out) {
    int t = blockIdx.x * 256 + threadIdx.x;
    if (t >= NG * NC) return;
    int g = t >> 1, cls = t & 1;
    float s = 0.f;
    const float* p = pooled + (size_t)g * DH;
    const float* w = Wfc + (size_t)cls * DH;
#pragma unroll 4
    for (int d = 0; d < DH; ++d) s += p[d] * w[d];
    float cnt = counts[g]; cnt = cnt < 1.f ? 1.f : cnt;
    float z = s / cnt + bfc[cls];
    out[g * NC + cls] = 1.f / (1.f + expf(-z));
}

// ---------------- workspace layout ----------------
#define ALIGN256(x) (((x) + 255) & ~(size_t)255)
static const size_t SZ_XBF   = (size_t)NN * DIN * 2;   // x in bf16
static const size_t SZ_AGG1F = (size_t)NN * DIN * 4;   // layer-1 agg f32
static const size_t SZ_AGG1B = (size_t)NN * DIN * 2;   // layer-1 agg bf16
static const size_t SZ_H1B   = (size_t)NN * DH * 2;    // h1 bf16
static const size_t SZ_AGG2F = (size_t)NN * DH * 4;    // layer-2 agg f32
static const size_t SZ_AGG2B = (size_t)NN * DH * 2;    // layer-2 agg bf16
static const size_t SZ_H2F   = (size_t)NN * DH * 4;    // h2 f32
static const size_t SZ_WP1   = (size_t)16 * 8 * 32 * 16 * 2;   // packed W layer1
static const size_t SZ_WP2   = (size_t)16 * 16 * 32 * 16 * 2;  // packed W layer2
static const size_t SZ_POOL  = (size_t)NG * DH * 4;
static const size_t SZ_CNT   = (size_t)NG * 4;

extern "C" void kernel_launch(void* const* d_in, const int* in_sizes, int n_in,
                              void* d_out, int out_size, void* d_ws, size_t ws_size,
                              hipStream_t stream) {
    const float*     x      = (const float*)d_in[0];
    const long long* ei     = (const long long*)d_in[1];
    const long long* batch  = (const long long*)d_in[2];
    const float*     W1root = (const float*)d_in[4];
    const float*     W1rel  = (const float*)d_in[5];
    const float*     b1     = (const float*)d_in[6];
    const float*     W2root = (const float*)d_in[7];
    const float*     W2rel  = (const float*)d_in[8];
    const float*     b2     = (const float*)d_in[9];
    const float*     Wfc    = (const float*)d_in[10];
    const float*     bfc    = (const float*)d_in[11];
    float*           out    = (float*)d_out;

    char* ws = (char*)d_ws;
    size_t o = 0;
    unsigned short* xbf   = (unsigned short*)(ws + o); o = ALIGN256(o + SZ_XBF);
    float*          agg1f = (float*)(ws + o);          o = ALIGN256(o + SZ_AGG1F);
    unsigned short* agg1b = (unsigned short*)(ws + o); o = ALIGN256(o + SZ_AGG1B);
    unsigned short* h1b   = (unsigned short*)(ws + o); o = ALIGN256(o + SZ_H1B);
    float*          agg2f = (float*)(ws + o);          o = ALIGN256(o + SZ_AGG2F);
    unsigned short* agg2b = (unsigned short*)(ws + o); o = ALIGN256(o + SZ_AGG2B);
    float*          h2f   = (float*)(ws + o);          o = ALIGN256(o + SZ_H2F);
    unsigned short* wp1   = (unsigned short*)(ws + o); o = ALIGN256(o + SZ_WP1);
    unsigned short* wp2   = (unsigned short*)(ws + o); o = ALIGN256(o + SZ_WP2);
    float*          pooled= (float*)(ws + o);          o = ALIGN256(o + SZ_POOL);
    float*          counts= (float*)(ws + o);          o = ALIGN256(o + SZ_CNT);

    // zero the accumulators (graph-capturable)
    hipMemsetAsync(agg1f, 0, SZ_AGG1F, stream);
    hipMemsetAsync(agg2f, 0, SZ_AGG2F, stream);
    hipMemsetAsync(pooled, 0, SZ_POOL, stream);
    hipMemsetAsync(counts, 0, SZ_CNT, stream);

    // pack weights into WMMA B-fragment layout
    k_pack<<<(16 * 8 * 512) / 256, 256, 0, stream>>>(W1root, W1rel, DIN, wp1);
    k_pack<<<(16 * 16 * 512) / 256, 256, 0, stream>>>(W2root, W2rel, DH, wp2);

    // x -> bf16
    k_f2bf<<<(NN * DIN) / 256, 256, 0, stream>>>(x, xbf, NN * DIN);

    // layer 1: aggregate raw features (f32 atomics), convert, fused GEMM+ReLU
    k_scatter_f32<<<(NE * 32) / 256, 256, 0, stream>>>(x, ei, agg1f, NE);
    k_f2bf<<<(NN * DIN) / 256, 256, 0, stream>>>(agg1f, agg1b, NN * DIN);
    k_gemm<false><<<NN / 16, 256, 0, stream>>>(xbf, agg1b, DIN, DIN, wp1, b1, (void*)h1b);

    // layer 2: aggregate h1 (bf16 gather -> f32 atomics), convert, fused GEMM+ReLU
    k_scatter_bf<<<(NE * 32) / 256, 256, 0, stream>>>(h1b, ei, agg2f, NE);
    k_f2bf<<<(NN * DH) / 256, 256, 0, stream>>>(agg2f, agg2b, NN * DH);
    k_gemm<true><<<NN / 16, 256, 0, stream>>>(h1b, agg2b, DH, DH, wp2, b2, (void*)h2f);

    // global mean pool + head
    k_counts<<<(NN + 255) / 256, 256, 0, stream>>>(batch, counts, NN);
    k_pool<<<(NN * 32) / 256, 256, 0, stream>>>(h2f, batch, pooled, NN);
    k_head<<<2, 256, 0, stream>>>(pooled, counts, Wfc, bfc, out);
}
